// _SharedPoolSparseExperts_25701084299293
// MI455X (gfx1250) — matrix operand order repaired and checked by reading, not compiled, not run on hardware
//
#include <hip/hip_runtime.h>
#include <math.h>
#include <stdint.h>

// ---------------- problem constants (from reference) ----------------
#define BB 4
#define NN 4096
#define CC 1024
#define EE 8
#define RR 64
#define TT (BB * NN)   // 16384 tokens
#define MT 16          // tokens per tile (WMMA M)

// packed-weight geometry: each 32x16 bf16 B-fragment stored as [32 lanes][16 halfs]
#define FRAG_HALFS (32 * 16)
#define WD_FRAGS 128   // per expert: 32 k-tiles x 4 n-tiles
#define WU_FRAGS 128   // per expert: 64 n-tiles x 2 k-tiles

// ---------------- CDNA5 WMMA types ----------------
typedef __attribute__((ext_vector_type(16))) __bf16 v16bf;
typedef __attribute__((ext_vector_type(8)))  float  v8f;

union FragU { uint4 q[2]; v16bf v; };

// round-to-nearest-even f32 -> bf16 bit pattern
__device__ __forceinline__ unsigned short f2bf(float f) {
  unsigned int u = __float_as_uint(f);
  u += 0x7FFFu + ((u >> 16) & 1u);
  return (unsigned short)(u >> 16);
}

// A-matrix 16x32 bf16 fragment (CDNA5 layout): lane L -> row M = L%16;
// lanes 0-15 hold K = k0+{0..7} & k0+16+{0..7}; lanes 16-31 K shifted by 8.
// Both runs are 16B-contiguous -> two ds_load_b128.
__device__ __forceinline__ v16bf load_a_frag_lds(const unsigned short* base,
                                                 int rowStride, int k0, int lane) {
  FragU r;
  int m  = lane & 15;
  int g8 = (lane >> 4) << 3;
  const unsigned short* p = base + m * rowStride + k0 + g8;
  r.q[0] = *(const uint4*)(p);
  r.q[1] = *(const uint4*)(p + 16);
  return r.v;
}

// B-fragment from prepacked global buffer: lane's 16 halfs are contiguous (32B)
// -> two global_load_b128.
__device__ __forceinline__ v16bf load_frag_packed(const unsigned short* pk,
                                                  int fragId, int lane) {
  FragU r;
  const uint4* q = (const uint4*)(pk + ((size_t)fragId * 32 + lane) * 16);
  r.q[0] = q[0];
  r.q[1] = q[1];
  return r.v;
}

// issue one per-lane async cache->LDS 16B transfer (GVS mode), ASYNCcnt-tracked
__device__ __forceinline__ void async_load_b128_to_lds(unsigned lds_addr,
                                                       unsigned voff,
                                                       const void* sbase) {
  asm volatile("global_load_async_to_lds_b128 %0, %1, %2"
               :: "v"(lds_addr), "v"(voff), "s"(sbase)
               : "memory");
}

__device__ __forceinline__ void wait_asynccnt0() {
#if __has_builtin(__builtin_amdgcn_s_wait_asynccnt)
  __builtin_amdgcn_s_wait_asynccnt(0);
#else
  asm volatile("s_wait_asynccnt 0x0" ::: "memory");
#endif
}

// ---------------- kernel 1: zero aux accumulators ----------------
__global__ void zero_acc_kernel(float* acc) {
  if (threadIdx.x < 2 * EE) acc[threadIdx.x] = 0.0f;
}

// ---------------- kernel 2: prepack Wd/Wu into bf16 fragment-major ----------
// One thread per (fragment, lane): writes 16 contiguous halfs.
// Wd frag id per expert: f = kt*4 + nt   (kt in [0,32), nt in [0,4),  B = Wd[e][C][R])
// Wu frag id per expert: f = nt*2 + kt   (nt in [0,64), kt in [0,2),  B = Wu[e][R][C])
__global__ void pack_weights_kernel(const float* __restrict__ Wd,
                                    const float* __restrict__ Wu,
                                    unsigned short* __restrict__ wdp,
                                    unsigned short* __restrict__ wup) {
  int idx  = blockIdx.x * blockDim.x + threadIdx.x;
  int lane = idx & 31;
  int fl   = idx >> 5;
  int n16  = lane & 15;
  int kadd = (lane >> 4) << 3;

  if (fl < EE * WD_FRAGS) {
    int e = fl >> 7, f = fl & 127;
    int kt = f >> 2, nt = f & 3;
    const float* src = Wd + ((size_t)e * CC + (size_t)(kt * 32 + kadd)) * RR + nt * 16 + n16;
    unsigned short* dst = wdp + ((size_t)fl * 32 + lane) * 16;
#pragma unroll
    for (int j = 0; j < 16; ++j) dst[j] = f2bf(src[(size_t)((j & 7) + ((j >> 3) << 4)) * RR]);
  } else if (fl < 2 * EE * WD_FRAGS) {
    int g = fl - EE * WD_FRAGS;
    int e = g >> 7, f = g & 127;
    int nt = f >> 1, kt = f & 1;
    const float* src = Wu + ((size_t)e * RR + (size_t)(kt * 32 + kadd)) * CC + nt * 16 + n16;
    unsigned short* dst = wup + ((size_t)g * 32 + lane) * 16;
#pragma unroll
    for (int j = 0; j < 16; ++j) dst[j] = f2bf(src[(size_t)((j & 7) + ((j >> 3) << 4)) * CC]);
  }
}

// ---------------- kernel 3: router + fused x -> bf16 conversion -------------
__global__ void router_kernel(const float* __restrict__ x,
                              const float* __restrict__ router_w,
                              const float* __restrict__ router_b,
                              const int*   __restrict__ key_id,
                              float* __restrict__ gates,
                              unsigned short* __restrict__ xb,
                              float* __restrict__ imp,
                              float* __restrict__ loadc) {
  int t = blockIdx.x * blockDim.x + threadIdx.x;
  if (t >= TT) return;
  int k = key_id[0];
  const float* rw = router_w + (size_t)k * CC * EE;  // [C][E]
  const float* rb = router_b + (size_t)k * EE;

  float logits[EE];
#pragma unroll
  for (int e = 0; e < EE; ++e) logits[e] = rb[e];

  const float* xr = x + (size_t)t * CC;
  unsigned short* xbr = xb + (size_t)t * CC;
  for (int c = 0; c < CC; ++c) {
    float xv = xr[c];
    xbr[c] = f2bf(xv);  // fused bf16 copy for the WMMA kernel
    const float* w = rw + (size_t)c * EE;
#pragma unroll
    for (int e = 0; e < EE; ++e) logits[e] = fmaf(xv, w[e], logits[e]);
  }

  float mx = logits[0];
#pragma unroll
  for (int e = 1; e < EE; ++e) mx = fmaxf(mx, logits[e]);
  float p[EE], s = 0.0f;
#pragma unroll
  for (int e = 0; e < EE; ++e) { p[e] = __expf(logits[e] - mx); s += p[e]; }
  float inv = 1.0f / s;
#pragma unroll
  for (int e = 0; e < EE; ++e) p[e] *= inv;

  int i1 = 0;
#pragma unroll
  for (int e = 1; e < EE; ++e) if (p[e] > p[i1]) i1 = e;
  int i2 = (i1 == 0) ? 1 : 0;
#pragma unroll
  for (int e = 0; e < EE; ++e) if (e != i1 && p[e] > p[i2]) i2 = e;

  float gsum = p[i1] + p[i2];
  float g1 = p[i1] / gsum, g2 = p[i2] / gsum;

  float* gr = gates + (size_t)t * EE;
#pragma unroll
  for (int e = 0; e < EE; ++e) gr[e] = 0.0f;
  gr[i1] = g1;
  gr[i2] = g2;

#pragma unroll
  for (int e = 0; e < EE; ++e) atomicAdd(&imp[e], p[e]);
  atomicAdd(&loadc[i1], 1.0f);
  atomicAdd(&loadc[i2], 1.0f);
}

// ---------------- kernel 4: expert FFN via WMMA ----------------
// One block per 16-token tile, 8 waves.
// Phase 1: wave w -> expert w: H = gelu(X @ Wd[e]) * gate  (bf16 to LDS)
// Phase 2: wave w -> output cols [w*128, w*128+128): out = sum_e H_e @ Wu[e]
__global__ __launch_bounds__(256) void ffn_kernel(const unsigned short* __restrict__ xb,
                                                  const unsigned short* __restrict__ wdp,
                                                  const unsigned short* __restrict__ wup,
                                                  const float* __restrict__ gates,
                                                  float* __restrict__ out) {
  __shared__ __align__(16) unsigned short Xs[MT * CC];       // 32 KB bf16 X tile
  __shared__ __align__(16) unsigned short Hs[EE * MT * RR];  // 16 KB gated bf16 H
  __shared__ float gs[MT * EE];
  __shared__ float eany[EE];

  const int t0   = blockIdx.x * MT;
  const int tid  = threadIdx.x;
  const int lane = tid & 31;
  const int wave = tid >> 5;

  // prefetch next tile's activations into L2 (global_prefetch_b8)
  if (t0 + MT < TT && tid == 0)
    __builtin_prefetch(xb + (size_t)(t0 + MT) * CC, 0, 1);

  // stage bf16 X tile: contiguous 32KB block, cache -> LDS via async engine
  // (no VGPR round-trip; tracked by ASYNCcnt)
  {
    const void* src = (const void*)(xb + (size_t)t0 * CC);
    unsigned ldsBase = (unsigned)(uintptr_t)(void*)Xs;  // low 32b = LDS offset
#pragma unroll
    for (int k = 0; k < (MT * CC * 2) / (256 * 16); ++k) {
      unsigned voff = (unsigned)(tid + k * 256) * 16u;
      async_load_b128_to_lds(ldsBase + voff, voff, src);
    }
  }
  for (int i = tid; i < MT * EE; i += 256) gs[i] = gates[(size_t)t0 * EE + i];
  wait_asynccnt0();
  __syncthreads();

  // ---------------- phase 1 ----------------
  {
    const int e = wave;
    float any = 0.0f;
#pragma unroll
    for (int m = 0; m < MT; ++m) any += gs[m * EE + e];
    if (lane == 0) eany[e] = any;

    const int n  = lane & 15;
    const int mh = (lane >> 4) << 3;

    if (any != 0.0f) {
      const unsigned short* wdpe = wdp + (size_t)e * WD_FRAGS * FRAG_HALFS;
      v8f acc[4] = {};
      for (int kt = 0; kt < 32; ++kt) {
        v16bf a = load_a_frag_lds(Xs, CC, kt * 32, lane);
#pragma unroll
        for (int t = 0; t < 4; ++t) {
          v16bf b = load_frag_packed(wdpe, kt * 4 + t, lane);
          acc[t] = __builtin_amdgcn_wmma_f32_16x16x32_bf16(
              false, a, false, b, (short)0, acc[t], false, false);
        }
      }
      // exact GELU, fold gate, store bf16 H
#pragma unroll
      for (int t = 0; t < 4; ++t) {
#pragma unroll
        for (int i = 0; i < 8; ++i) {
          int m = i + mh;
          float h = acc[t][i];
          h = 0.5f * h * (1.0f + erff(h * 0.70710678118654752f));
          h *= gs[m * EE + e];
          Hs[(e * MT + m) * RR + t * 16 + n] = f2bf(h);
        }
      }
    } else {
#pragma unroll
      for (int t = 0; t < 4; ++t)
#pragma unroll
        for (int i = 0; i < 8; ++i)
          Hs[(e * MT + (i + mh)) * RR + t * 16 + n] = 0;
    }
  }
  __syncthreads();

  // ---------------- phase 2 ----------------
  {
    const int c_base = wave * 128;
    const int nt0 = wave * 8;  // first 16-col tile index owned by this wave
    v8f acc[8] = {};
    for (int e = 0; e < EE; ++e) {
      if (eany[e] == 0.0f) continue;  // gated H is zero; skip compute
      const unsigned short* wupe = wup + (size_t)e * WU_FRAGS * FRAG_HALFS;
      v16bf a0 = load_a_frag_lds(Hs + e * MT * RR, RR, 0, lane);
      v16bf a1 = load_a_frag_lds(Hs + e * MT * RR, RR, 32, lane);
#pragma unroll
      for (int t = 0; t < 8; ++t) {
        v16bf b0 = load_frag_packed(wupe, (nt0 + t) * 2 + 0, lane);
        acc[t] = __builtin_amdgcn_wmma_f32_16x16x32_bf16(
            false, a0, false, b0, (short)0, acc[t], false, false);
        v16bf b1 = load_frag_packed(wupe, (nt0 + t) * 2 + 1, lane);
        acc[t] = __builtin_amdgcn_wmma_f32_16x16x32_bf16(
            false, a1, false, b1, (short)0, acc[t], false, false);
      }
    }
    const int n  = lane & 15;
    const int mh = (lane >> 4) << 3;
#pragma unroll
    for (int t = 0; t < 8; ++t)
#pragma unroll
      for (int i = 0; i < 8; ++i) {
        int m = i + mh;
        out[(size_t)(t0 + m) * CC + c_base + t * 16 + n] = acc[t][i];
      }
  }
}

// ---------------- kernel 5: aux loss ----------------
__global__ void finalize_kernel(const float* __restrict__ imp,
                                const float* __restrict__ loadc,
                                float* __restrict__ out_aux) {
  if (threadIdx.x == 0 && blockIdx.x == 0) {
    float s = 0.0f;
#pragma unroll
    for (int e = 0; e < EE; ++e) {
      float importance = imp[e] / (float)TT;
      float load = (loadc[e] / (float)TT) * 0.5f;  // / TOP_K
      s += importance * load;
    }
    out_aux[0] = 0.001f * (float)EE * s;
  }
}

// ---------------- host side ----------------
extern "C" void kernel_launch(void* const* d_in, const int* in_sizes, int n_in,
                              void* d_out, int out_size, void* d_ws, size_t ws_size,
                              hipStream_t stream) {
  (void)in_sizes; (void)n_in; (void)out_size; (void)ws_size;

  const float* x        = (const float*)d_in[0];
  const float* router_w = (const float*)d_in[1];
  const float* router_b = (const float*)d_in[2];
  const float* Wd       = (const float*)d_in[3];
  const float* Wu       = (const float*)d_in[4];
  const int*   key_id   = (const int*)d_in[5];

  float* out = (float*)d_out;

  // workspace layout (all offsets 16B-aligned):
  //   xb   : bf16 x              [TT*CC]            32 MB
  //   wdp  : packed bf16 Wd      [EE*128*512]        1 MB
  //   wup  : packed bf16 Wu      [EE*128*512]        1 MB
  //   gates: f32                 [TT*EE]           512 KB
  //   imp/loadc: f32             [2*EE]              64 B
  unsigned short* xb  = (unsigned short*)d_ws;
  unsigned short* wdp = xb + (size_t)TT * CC;
  unsigned short* wup = wdp + (size_t)EE * WD_FRAGS * FRAG_HALFS;
  float* gates = (float*)(wup + (size_t)EE * WU_FRAGS * FRAG_HALFS);
  float* imp   = gates + (size_t)TT * EE;
  float* loadc = imp + EE;

  zero_acc_kernel<<<1, 32, 0, stream>>>(imp);
  pack_weights_kernel<<<(2 * EE * WD_FRAGS * 32) / 256, 256, 0, stream>>>(Wd, Wu, wdp, wup);
  router_kernel<<<TT / 256, 256, 0, stream>>>(x, router_w, router_b, key_id,
                                              gates, xb, imp, loadc);
  ffn_kernel<<<TT / MT, 256, 0, stream>>>(xb, wdp, wup, gates, out);
  finalize_kernel<<<1, 32, 0, stream>>>(imp, loadc, out + (size_t)TT * CC);
}
